// QRNN_78889959293551
// MI455X (gfx1250) — matrix operand order repaired and checked by reading, not compiled, not run on hardware
//
#include <hip/hip_runtime.h>
#include <hip/hip_bf16.h>
#include <cstdint>
#include <cstddef>

typedef __attribute__((ext_vector_type(16))) __bf16         v16bf;
typedef __attribute__((ext_vector_type(8)))  float          v8f;
typedef __attribute__((ext_vector_type(4)))  unsigned int   v4u;
typedef __attribute__((ext_vector_type(4)))  unsigned short v4us;

#define B_  32
#define T_  1024
#define D_  512
#define U_  512
#define TT  1022           /* T-2 */
#define K_  1536           /* 3*D */
#define N_  1536           /* 3*U */
#define M_  (B_*TT)        /* 32704 */
#define SA  48             /* padded LDS row stride in halves (96B rows, 16B-aligned chunks) */
#define NK  (K_/32)        /* 48 K-steps */

__device__ __forceinline__ unsigned short f2bf(float f) {
    unsigned int u = __float_as_uint(f);
    u += 0x7FFFu + ((u >> 16) & 1u);          // round-to-nearest-even
    return (unsigned short)(u >> 16);
}
__device__ __forceinline__ float bf2f(unsigned short h) {
    return __uint_as_float(((unsigned int)h) << 16);
}
__device__ __forceinline__ float sigmoidf_(float x) {
    return 1.0f / (1.0f + __expf(-x));
}

// Async global->LDS copy of 16B per lane (ASYNCcnt-tracked, bypasses VGPRs).
// Generic pointers to __shared__ carry the LDS offset in their low 32 bits
// (flat->LDS aperture mapping truncates to addr[31:0]).
__device__ __forceinline__ void async_b128(unsigned ldsOff, const void* gp) {
    asm volatile("global_load_async_to_lds_b128 %0, %1, off"
                 :: "v"(ldsOff), "v"((unsigned long long)(size_t)gp)
                 : "memory");
}
__device__ __forceinline__ void wait_async0() {
    asm volatile("s_wait_asynccnt 0x0" ::: "memory");
}

// ---------------- kernel 1: x f32 -> bf16 ----------------
__global__ void cvt_x(const float4* __restrict__ x, v4us* __restrict__ out, int n4) {
    int i = blockIdx.x * blockDim.x + threadIdx.x;
    if (i < n4) {
        float4 v = x[i];
        v4us r = { f2bf(v.x), f2bf(v.y), f2bf(v.z), f2bf(v.w) };
        out[i] = r;
    }
}

// ------- kernel 2: weights (K=3*D, N=3*U) f32 -> bf16 transposed [N][K] -------
__global__ void cvt_wT(const float* __restrict__ w, unsigned short* __restrict__ bt, int total) {
    int i = blockIdx.x * blockDim.x + threadIdx.x;
    if (i < total) {
        int n = i / K_;
        int k = i - n * K_;
        bt[i] = f2bf(w[(size_t)k * N_ + n]);   // bt[n*K + k] = w[k][n]
    }
}

// ---------------- kernel 3: implicit-GEMM conv + fused activations ----------------
// Block tile 64(M) x 256(N); 8 waves in 2x4, each wave 32x64 via 8 WMMAs/K-step.
// Double-buffered LDS fed by global_load_async_to_lds_b128 (copy overlaps WMMA).
__global__ __launch_bounds__(256) void qrnn_gemm(
    const unsigned short* __restrict__ xbf,   // (B, T, D) bf16
    const unsigned short* __restrict__ wT,    // (N, K) bf16
    unsigned short* __restrict__ gates)       // (B, TT, 3U) bf16, activated
{
    __shared__ unsigned short As[2][64 * SA];
    __shared__ unsigned short Bs[2][256 * SA];

    const int tid  = threadIdx.x;
    const int lane = tid & 31;
    const int wave = tid >> 5;
    const int m0 = blockIdx.x * 64;
    const int n0 = blockIdx.y * 256;

    // ---- A staging: thread -> (row = tid/4, 16B chunk = tid%4). Row m -> (b,t).
    const int arr = tid >> 2;
    const int ach = tid & 3;
    const int am  = m0 + arr;
    const int ab  = am / TT;
    const int at  = am - ab * TT;
    const unsigned short* aSrc = xbf + (size_t)ab * (T_ * D_) + (size_t)at * D_ + ach * 8;
    const unsigned aOff[2] = { (unsigned)(size_t)&As[0][arr * SA + ach * 8],
                               (unsigned)(size_t)&As[1][arr * SA + ach * 8] };

    // ---- B staging: thread -> row n = tid (256 rows), 4 x 16B chunks (32 halves).
    const unsigned short* bSrc = wT + (size_t)(n0 + tid) * K_;
    const unsigned bOff[2] = { (unsigned)(size_t)&Bs[0][tid * SA],
                               (unsigned)(size_t)&Bs[1][tid * SA] };

    // ---- wave tiling: wm in {0,1} -> 32 M-rows, wn in {0..3} -> 64 N-cols
    const int wm   = wave >> 2;
    const int wn   = wave & 3;
    const int half = lane >> 4;     // 0/1
    const int l16  = lane & 15;

    // fragment LDS pointers in buffer 0 (ISA 7.12.2 16-bit layouts)
    const unsigned short* aF = &As[0][(wm * 32 + l16) * SA + half * 8];   // +16*SA for mt=1
    const unsigned short* bF = &Bs[0][(wn * 64 + l16) * SA + half * 8];   // +16*SA per nt

    v8f acc[2][4];
#pragma unroll
    for (int mt = 0; mt < 2; ++mt)
#pragma unroll
        for (int nt = 0; nt < 4; ++nt) acc[mt][nt] = (v8f){};

    // prologue: request K-step 0 into buffer 0
    {
        async_b128(aOff[0], aSrc);
        const unsigned short* bg = bSrc;
        async_b128(bOff[0],      bg);
        async_b128(bOff[0] + 16, bg + 8);
        async_b128(bOff[0] + 32, bg + 16);
        async_b128(bOff[0] + 48, bg + 24);
    }

    for (int kt = 0; kt < NK; ++kt) {
        const int cur = kt & 1;
        wait_async0();            // our async writes for step kt have landed in LDS
        __syncthreads();          // everyone's writes for step kt are visible

        if (kt + 1 < NK) {        // overlap: request kt+1 into the other buffer
            const int nxt = cur ^ 1;
            async_b128(aOff[nxt], aSrc + (kt + 1) * 32);
            const unsigned short* bg = bSrc + (kt + 1) * 32;
            async_b128(bOff[nxt],      bg);
            async_b128(bOff[nxt] + 16, bg + 8);
            async_b128(bOff[nxt] + 32, bg + 16);
            async_b128(bOff[nxt] + 48, bg + 24);
        }

        const unsigned short* aP = aF + cur * (64 * SA);
        const unsigned short* bP = bF + cur * (256 * SA);

        union Frag { v4u q[2]; v16bf v; } fa[2], fb[4];
#pragma unroll
        for (int mt = 0; mt < 2; ++mt) {
            const unsigned short* p = aP + mt * (16 * SA);
            fa[mt].q[0] = *(const v4u*)p;
            fa[mt].q[1] = *(const v4u*)(p + 16);
        }
#pragma unroll
        for (int nt = 0; nt < 4; ++nt) {
            const unsigned short* p = bP + nt * (16 * SA);
            fb[nt].q[0] = *(const v4u*)p;
            fb[nt].q[1] = *(const v4u*)(p + 16);
        }
#pragma unroll
        for (int mt = 0; mt < 2; ++mt)
#pragma unroll
            for (int nt = 0; nt < 4; ++nt)
                acc[mt][nt] = __builtin_amdgcn_wmma_f32_16x16x32_bf16(
                    false, fa[mt].v, false, fb[nt].v, (short)0, acc[mt][nt], false, false);

        __syncthreads();          // reads of buffer `cur` done before it is re-filled
    }

    // ---- store: fused activation, bf16 gates. C layout: VGPR r -> M = r + 8*half, N = l16.
    const int mBase = m0 + wm * 32;
    const int nBase = n0 + wn * 64;
#pragma unroll
    for (int mt = 0; mt < 2; ++mt) {
#pragma unroll
        for (int r = 0; r < 8; ++r) {
            const int m = mBase + mt * 16 + r + half * 8;   // m = b*TT + t already
            const size_t rowOff = (size_t)m * N_;
#pragma unroll
            for (int nt = 0; nt < 4; ++nt) {
                const int col = nBase + nt * 16 + l16;
                float v = acc[mt][nt][r];
                v = (col < U_) ? tanhf(v) : sigmoidf_(v);
                gates[rowOff + col] = f2bf(v);
            }
        }
    }
}

// ---------------- kernel 4: sequential scan over t, parallel over (b,u) ----------------
__global__ void qrnn_scan(const unsigned short* __restrict__ gates, float* __restrict__ out) {
    const int id = blockIdx.x * blockDim.x + threadIdx.x;   // 0 .. B_*U_-1
    const int b = id >> 9;
    const int u = id & 511;
    const unsigned short* g = gates + (size_t)b * TT * N_ + u;
    float* o = out + (size_t)b * TT * U_ + u;

    float c = 0.0f;
    float z = bf2f(g[0]), f = bf2f(g[U_]), og = bf2f(g[2 * U_]);
    for (int t = 0; t < TT; ++t) {
        float zn = 0.0f, fn = 0.0f, on = 0.0f;
        if (t + 1 < TT) {
            const unsigned short* gn = g + (size_t)(t + 1) * N_;
            zn = bf2f(gn[0]); fn = bf2f(gn[U_]); on = bf2f(gn[2 * U_]);
        }
        c = f * c + (1.0f - f) * z;
        o[(size_t)t * U_] = og * c;
        z = zn; f = fn; og = on;
    }
}

extern "C" void kernel_launch(void* const* d_in, const int* in_sizes, int n_in,
                              void* d_out, int out_size, void* d_ws, size_t ws_size,
                              hipStream_t stream) {
    const float* x = (const float*)d_in[0];       // (32,1024,512) f32
    const float* w = (const float*)d_in[1];       // (3,512,1536)  f32
    float* out = (float*)d_out;                   // (32,1022,512) f32

    char* ws = (char*)d_ws;
    unsigned short* xbf   = (unsigned short*)ws;                                  // 33,554,432 B
    unsigned short* wT    = (unsigned short*)(ws + 33554432);                     //  4,718,592 B
    unsigned short* gates = (unsigned short*)(ws + 33554432 + 4718592);           // 100,466,688 B

    cvt_x <<<16384, 256, 0, stream>>>((const float4*)x, (v4us*)xbf, 4194304);
    cvt_wT<<<9216,  256, 0, stream>>>(w, wT, K_ * N_);

    dim3 grid(M_ / 64, N_ / 256);                 // 511 x 6
    qrnn_gemm<<<grid, 256, 0, stream>>>(xbf, wT, gates);

    qrnn_scan<<<(B_ * U_) / 256, 256, 0, stream>>>(gates, out);
}
